// MySelfAttention_23012434772025
// MI455X (gfx1250) — matrix operand order repaired and checked by reading, not compiled, not run on hardware
//
#include <hip/hip_runtime.h>
#include <math.h>

typedef __attribute__((ext_vector_type(8)))  _Float16 v8h;
typedef __attribute__((ext_vector_type(16))) _Float16 v16h;
typedef __attribute__((ext_vector_type(8)))  float    v8f;

#define DD 1024
#define TT 2048
#define BB 4

// ---------------- WMMA helpers ----------------

__device__ __forceinline__ v8f wmma_f16(v16h a, v16h b, v8f c) {
  // v_wmma_f32_16x16x32_f16: (neg_a, A, neg_b, B, c_mod, C, reuse_a, reuse_b)
  return __builtin_amdgcn_wmma_f32_16x16x32_f16(false, a, false, b, (short)0, c,
                                                false, false);
}

// A-matrix fragment (16x32 f16, M x K). Layout: lanes 0-15 hold M=lane,
// K = 0..15; lanes 16-31 hold K = 16..31. One 32B load.
__device__ __forceinline__ v16h frag_a(const _Float16* __restrict__ base,
                                       int ld, int row, int kb) {
  int lane = threadIdx.x & 31;
  const _Float16* p =
      base + (size_t)(row + (lane & 15)) * ld + kb + ((lane >> 4) << 4);
  return *(const v16h*)p;
}

// B-matrix fragment (32x16 f16, K x N), element B[k][n] = src[row0+n][kb+k]
// (src rows contiguous along K). Layout as in frag_a: lanes 0-15 hold K=0..15,
// lanes 16-31 hold K=16..31, column N = lane&15. One 32B load per lane.
__device__ __forceinline__ v16h frag_b(const _Float16* __restrict__ base,
                                       int ld, int row, int kb) {
  int lane = threadIdx.x & 31;
  const _Float16* p =
      base + (size_t)(row + (lane & 15)) * ld + kb + ((lane >> 4) << 4);
  return *(const v16h*)p;
}

// ---------------- Kernels ----------------

// f32 -> f16 conversion
__global__ void k_cvt(const float* __restrict__ src, _Float16* __restrict__ dst,
                      int n) {
  int i = blockIdx.x * blockDim.x + threadIdx.x;
  if (i < n) dst[i] = (_Float16)src[i];
}

// C[m, n] = sum_k X[m,k] * W[n,k] + bias[n]
// transposed==0 : out is f16 [Mtot, D]
// transposed==1 : out is f16 [B][D][T]  (V stored feature-major for PV GEMM)
// Block: 128 thr = 4 waves; block tile 128(M)x64(N); wave tile 64x32
// (8 accums = 64 VGPRs -> no spills, 0.75 fragment loads per WMMA).
__global__ void __launch_bounds__(128)
k_qkv_gemm(const _Float16* __restrict__ X, const _Float16* __restrict__ W,
           const float* __restrict__ bias, _Float16* __restrict__ out,
           int transposed) {
  int w = threadIdx.x >> 5, lane = threadIdx.x & 31;
  int m0 = blockIdx.x * 128 + (w & 1) * 64;
  int n0 = blockIdx.y * 64 + (w >> 1) * 32;
  v8f c[4][2] = {};
#pragma unroll 2
  for (int k = 0; k < DD; k += 32) {
    v16h a[4], b0, b1;
#pragma unroll
    for (int i = 0; i < 4; ++i) a[i] = frag_a(X, DD, m0 + 16 * i, k);
    b0 = frag_b(W, DD, n0, k);
    b1 = frag_b(W, DD, n0 + 16, k);
#pragma unroll
    for (int mi = 0; mi < 4; ++mi) {
      c[mi][0] = wmma_f16(a[mi], b0, c[mi][0]);
      c[mi][1] = wmma_f16(a[mi], b1, c[mi][1]);
    }
  }
  int hi = lane >> 4, cl = lane & 15;
#pragma unroll
  for (int mi = 0; mi < 4; ++mi)
#pragma unroll
    for (int ni = 0; ni < 2; ++ni) {
      int cb = n0 + ni * 16 + cl;
      float bv = bias[cb];
#pragma unroll
      for (int v = 0; v < 8; ++v) {
        int rm = m0 + mi * 16 + v + 8 * hi;
        float val = c[mi][ni][v] + bv;
        if (!transposed)
          out[(size_t)rm * DD + cb] = (_Float16)val;
        else  // batch = rm / T, token = rm % T
          out[((size_t)(rm >> 11) * DD + cb) * TT + (rm & (TT - 1))] =
              (_Float16)val;
      }
    }
}

// Scores + online softmax stats. Block (4 waves) handles 64 query rows of one
// batch, loops over key blocks of 128. Wave w covers key cols [w*32, w*32+32)
// for all 64 query rows (8 accums). S (f16) written to global; row max m and
// row sumexp l accumulated online via an LDS-staged f32 tile.
__global__ void __launch_bounds__(128)
k_attn_stats(const _Float16* __restrict__ Q, const _Float16* __restrict__ K,
             _Float16* __restrict__ S, float* __restrict__ mout,
             float* __restrict__ lout) {
  __shared__ float sbuf[64 * 128];  // 32 KB
  int b = blockIdx.z;
  int q0 = blockIdx.x * 64;
  int w = threadIdx.x >> 5, lane = threadIdx.x & 31;
  int nh = w * 32;  // this wave's 32 key columns within the 128-key block
  int hi = lane >> 4, cl = lane & 15;
  const _Float16* Qb = Q + (size_t)b * TT * DD;
  const _Float16* Kb = K + (size_t)b * TT * DD;
  _Float16* Sb = S + (size_t)b * TT * TT;
  float mrun = -3.0e38f, lrun = 0.0f;
  const float sc = 0.03125f;  // 1/sqrt(1024)

  for (int kb = 0; kb < TT; kb += 128) {
    v8f c[4][2] = {};
#pragma unroll 2
    for (int k = 0; k < DD; k += 32) {
      v16h a[4], b0, b1;
#pragma unroll
      for (int i = 0; i < 4; ++i) a[i] = frag_a(Qb, DD, q0 + 16 * i, k);
      b0 = frag_b(Kb, DD, kb + nh, k);
      b1 = frag_b(Kb, DD, kb + nh + 16, k);
#pragma unroll
      for (int mi = 0; mi < 4; ++mi) {
        c[mi][0] = wmma_f16(a[mi], b0, c[mi][0]);
        c[mi][1] = wmma_f16(a[mi], b1, c[mi][1]);
      }
    }
#pragma unroll
    for (int mi = 0; mi < 4; ++mi)
#pragma unroll
      for (int ni = 0; ni < 2; ++ni) {
#pragma unroll
        for (int v = 0; v < 8; ++v) {
          int r = mi * 16 + v + 8 * hi;
          int cc = nh + ni * 16 + cl;
          float s = c[mi][ni][v] * sc;
          sbuf[r * 128 + cc] = s;
          Sb[(size_t)(q0 + r) * TT + kb + cc] = (_Float16)s;
        }
      }
    __syncthreads();
    if (threadIdx.x < 64) {
      int r = threadIdx.x;
      float bm = -3.0e38f;
      for (int cc = 0; cc < 128; ++cc) bm = fmaxf(bm, sbuf[r * 128 + cc]);
      float mn = fmaxf(mrun, bm);
      float acc = 0.0f;
      for (int cc = 0; cc < 128; ++cc) acc += expf(sbuf[r * 128 + cc] - mn);
      lrun = lrun * expf(mrun - mn) + acc;
      mrun = mn;
    }
    __syncthreads();
  }
  if (threadIdx.x < 64) {
    mout[(size_t)b * TT + q0 + threadIdx.x] = mrun;
    lout[(size_t)b * TT + q0 + threadIdx.x] = lrun;
  }
}

// P = exp(S - m_row) / l_row, f16 in place.
__global__ void k_softmax_norm(_Float16* __restrict__ S,
                               const float* __restrict__ m,
                               const float* __restrict__ l) {
  size_t idx = (size_t)blockIdx.x * blockDim.x + threadIdx.x;
  size_t row = idx >> 11;  // (b*T + t), since T == 2048
  float s = (float)S[idx];
  S[idx] = (_Float16)(expf(s - m[row]) / l[row]);
}

// O[b,t,d] = sum_s P[b,t,s] * V[b,s,d], with V stored transposed Vt[b][d][s].
// Block: 4 waves; block tile 128(M)x64(N); wave tile 64x32.
__global__ void __launch_bounds__(128)
k_pv_gemm(const _Float16* __restrict__ P, const _Float16* __restrict__ Vt,
          float* __restrict__ out) {
  int b = blockIdx.z;
  int w = threadIdx.x >> 5, lane = threadIdx.x & 31;
  int m0 = blockIdx.x * 128 + (w & 1) * 64;
  int n0 = blockIdx.y * 64 + (w >> 1) * 32;
  const _Float16* Pb = P + (size_t)b * TT * TT;
  const _Float16* Vb = Vt + (size_t)b * DD * TT;
  float* Ob = out + (size_t)b * TT * DD;
  v8f c[4][2] = {};
#pragma unroll 2
  for (int k = 0; k < TT; k += 32) {
    v16h a[4], b0, b1;
#pragma unroll
    for (int i = 0; i < 4; ++i) a[i] = frag_a(Pb, TT, m0 + 16 * i, k);
    b0 = frag_b(Vb, TT, n0, k);
    b1 = frag_b(Vb, TT, n0 + 16, k);
#pragma unroll
    for (int mi = 0; mi < 4; ++mi) {
      c[mi][0] = wmma_f16(a[mi], b0, c[mi][0]);
      c[mi][1] = wmma_f16(a[mi], b1, c[mi][1]);
    }
  }
  int hi = lane >> 4, cl = lane & 15;
#pragma unroll
  for (int mi = 0; mi < 4; ++mi)
#pragma unroll
    for (int ni = 0; ni < 2; ++ni) {
      int cb = n0 + ni * 16 + cl;
#pragma unroll
      for (int v = 0; v < 8; ++v) {
        int rm = m0 + mi * 16 + v + 8 * hi;
        Ob[(size_t)rm * DD + cb] = c[mi][ni][v];
      }
    }
}

// ---------------- Launch ----------------

extern "C" void kernel_launch(void* const* d_in, const int* in_sizes, int n_in,
                              void* d_out, int out_size, void* d_ws,
                              size_t ws_size, hipStream_t stream) {
  (void)in_sizes; (void)n_in; (void)out_size; (void)ws_size;
  const float* x  = (const float*)d_in[0];
  const float* wq = (const float*)d_in[1];
  const float* bq = (const float*)d_in[2];
  const float* wk = (const float*)d_in[3];
  const float* bk = (const float*)d_in[4];
  const float* wv = (const float*)d_in[5];
  const float* bv = (const float*)d_in[6];
  float* out = (float*)d_out;

  char* ws = (char*)d_ws;
  const size_t NX = (size_t)BB * TT * DD;  // 8M elements
  const size_t NW = (size_t)DD * DD;       // 1M elements
  size_t off = 0;
  _Float16* xh  = (_Float16*)(ws + off); off += NX * 2;           // 16 MB
  _Float16* wqh = (_Float16*)(ws + off); off += NW * 2;           //  2 MB
  _Float16* wkh = (_Float16*)(ws + off); off += NW * 2;
  _Float16* wvh = (_Float16*)(ws + off); off += NW * 2;
  _Float16* Qh  = (_Float16*)(ws + off); off += NX * 2;           // 16 MB
  _Float16* Kh  = (_Float16*)(ws + off); off += NX * 2;           // 16 MB
  _Float16* Vth = (_Float16*)(ws + off); off += NX * 2;           // 16 MB
  _Float16* S   = (_Float16*)(ws + off); off += (size_t)BB * TT * TT * 2; // 32MB
  float* mbuf   = (float*)(ws + off); off += (size_t)BB * TT * 4;
  float* lbuf   = (float*)(ws + off); off += (size_t)BB * TT * 4;

  // 1) f32 -> f16 staging
  k_cvt<<<(int)(NX / 256), 256, 0, stream>>>(x, xh, (int)NX);
  k_cvt<<<(int)(NW / 256), 256, 0, stream>>>(wq, wqh, (int)NW);
  k_cvt<<<(int)(NW / 256), 256, 0, stream>>>(wk, wkh, (int)NW);
  k_cvt<<<(int)(NW / 256), 256, 0, stream>>>(wv, wvh, (int)NW);

  // 2) QKV projections (WMMA f16). V written feature-major per batch.
  dim3 gq(BB * TT / 128, DD / 64);
  k_qkv_gemm<<<gq, 128, 0, stream>>>(xh, wqh, bq, Qh, 0);
  k_qkv_gemm<<<gq, 128, 0, stream>>>(xh, wkh, bk, Kh, 0);
  k_qkv_gemm<<<gq, 128, 0, stream>>>(xh, wvh, bv, Vth, 1);

  // 3) Scores (WMMA) + online row max / sumexp
  dim3 gs(TT / 64, 1, BB);
  k_attn_stats<<<gs, 128, 0, stream>>>(Qh, Kh, S, mbuf, lbuf);

  // 4) Normalize: P = exp(S - m) / l, in place
  k_softmax_norm<<<(int)((size_t)BB * TT * TT / 256), 256, 0, stream>>>(S, mbuf,
                                                                        lbuf);

  // 5) O = P @ V  (WMMA f16, f32 output)
  dim3 gp(TT / 128, DD / 64, BB);
  k_pv_gemm<<<gp, 128, 0, stream>>>(S, Vth, out);
}
